// sLSTMCell_747324309576
// MI455X (gfx1250) — matrix operand order repaired and checked by reading, not compiled, not run on hardware
//
#include <hip/hip_runtime.h>
#include <hip/hip_fp16.h>

typedef __attribute__((ext_vector_type(16))) _Float16 v16h;
typedef __attribute__((ext_vector_type(8)))  _Float16 v8h;
typedef __attribute__((ext_vector_type(2)))  __fp16   fp16x2;   // matches cvt_pkrtz return
typedef __attribute__((ext_vector_type(4)))  __fp16   fp16x4;
typedef __attribute__((ext_vector_type(8)))  float    v8f;
typedef __attribute__((ext_vector_type(4)))  int      v4i;

typedef __attribute__((address_space(1))) v4i g_v4i;   // global int4
typedef __attribute__((address_space(3))) v4i l_v4i;   // LDS int4

#define B_SZ 4096
#define I_SZ 2048
#define H_SZ 2048
#define K_SZ 4096              // I + H
#define N_SZ 8192              // 4H
#define SLOT ((size_t)B_SZ * (size_t)H_SZ)

#define BM 128
#define BN 128
#define BK 32
#define LDA 40                 // padded halves per LDS row (80B = 20 banks, conflict-free)

// ------------------------------------------------------------------
// Kernel 0a: W f32 [K][N]  ->  Wt f16 [N][K]  (convert + transpose)
// ------------------------------------------------------------------
__global__ __launch_bounds__(256) void convW(const float* __restrict__ W,
                                             _Float16* __restrict__ Wt) {
  __shared__ _Float16 t[64 * 72];           // 72-half row pitch: 144B, 16B-aligned rows
  const int n0 = blockIdx.x * 64;
  const int k0 = blockIdx.y * 64;
  const int tid = threadIdx.x;
#pragma unroll
  for (int i = 0; i < 4; ++i) {
    int idx = tid + i * 256;
    int kr = idx >> 4;                      // 0..63
    int nc = (idx & 15) << 2;               // 0..60
    const float4 v = *(const float4*)(W + (size_t)(k0 + kr) * N_SZ + n0 + nc);
    t[(nc + 0) * 72 + kr] = (_Float16)v.x;
    t[(nc + 1) * 72 + kr] = (_Float16)v.y;
    t[(nc + 2) * 72 + kr] = (_Float16)v.z;
    t[(nc + 3) * 72 + kr] = (_Float16)v.w;
  }
  __syncthreads();
#pragma unroll
  for (int i = 0; i < 2; ++i) {
    int idx = tid + i * 256;
    int nr = idx >> 3;                      // 0..63
    int kc = (idx & 7) << 3;                // 0..56
    uint4 d = *(const uint4*)(&t[nr * 72 + kc]);
    *(uint4*)(Wt + (size_t)(n0 + nr) * K_SZ + k0 + kc) = d;
  }
}

// ------------------------------------------------------------------
// Kernel 0b: Ah f16 [B][K] = concat(x, h) converted (row-major)
// ------------------------------------------------------------------
__global__ __launch_bounds__(256) void convA(const float* __restrict__ x,
                                             const float* __restrict__ h,
                                             _Float16* __restrict__ Ah) {
  const size_t e = ((size_t)blockIdx.x * 256 + threadIdx.x) * 4;   // element index
  const int row = (int)(e >> 12);           // / 4096
  const int col = (int)(e & 4095);
  const float* src = (col < I_SZ) ? (x + (size_t)row * I_SZ + col)
                                  : (h + (size_t)row * H_SZ + (col - I_SZ));
  const float4 v = *(const float4*)src;
  fp16x2 p0 = __builtin_amdgcn_cvt_pkrtz(v.x, v.y);
  fp16x2 p1 = __builtin_amdgcn_cvt_pkrtz(v.z, v.w);
  *(fp16x4*)(Ah + e) = __builtin_shufflevector(p0, p1, 0, 1, 2, 3);
}

// ------------------------------------------------------------------
// Kernel 1: gates = Ah @ Wt^T + Bias, scattered into d_out gate slots.
// Tiles move global->LDS via GLOBAL_LOAD_ASYNC_TO_LDS_B128 (no VGPR staging).
// Uniform, unroll-1 loop body with XOR'd LDS offsets -> stable acc registers.
// ------------------------------------------------------------------
__global__ __launch_bounds__(256, 1) void gemm_gates(
    const _Float16* __restrict__ Ah, const _Float16* __restrict__ Wt,
    const float* __restrict__ bias, float* __restrict__ out) {
  __shared__ _Float16 ldsA[2 * BM * LDA];
  __shared__ _Float16 ldsB[2 * BN * LDA];
  const unsigned ABUF = BM * LDA;           // halves per A buffer
  const unsigned BBUF = BN * LDA;           // halves per B buffer

  const int tid   = threadIdx.x;
  const int lane  = tid & 31;
  const int wave  = tid >> 5;
  const int waveM = wave & 3;               // 4 waves along M -> 32 rows each
  const int waveN = wave >> 2;              // 2 waves along N -> 64 cols each
  const int l15   = lane & 15;
  const int lHi   = lane >> 4;

  const int bm = blockIdx.y * BM;
  const int bn = blockIdx.x * BN;

  const v8f zero8 = {0.f, 0.f, 0.f, 0.f, 0.f, 0.f, 0.f, 0.f};
  v8f acc[2][4];
#pragma unroll
  for (int mt = 0; mt < 2; ++mt)
#pragma unroll
    for (int nt = 0; nt < 4; ++nt) acc[mt][nt] = zero8;

  // Per-thread copy slice: 2 x (row r, 8 halves at kk) for each tile
  const int cr  = tid >> 2;                 // 0..63 (+64 on second pass)
  const int ckk = (tid & 3) << 3;           // 0,8,16,24

  auto asyncTiles = [&](int kt, unsigned dA, unsigned dB) {
    const int kb = kt * BK;
#pragma unroll
    for (int i = 0; i < 2; ++i) {
      const int r = cr + i * 64;
      __builtin_amdgcn_global_load_async_to_lds_b128(
          (g_v4i*)(Ah + (size_t)(bm + r) * K_SZ + kb + ckk),
          (l_v4i*)(&ldsA[dA + r * LDA + ckk]), 0, 0);
      __builtin_amdgcn_global_load_async_to_lds_b128(
          (g_v4i*)(Wt + (size_t)(bn + r) * K_SZ + kb + ckk),
          (l_v4i*)(&ldsB[dB + r * LDA + ckk]), 0, 0);
    }
  };

  auto compute = [&](unsigned oA, unsigned oB) {
    // A frags (16x32, ISA layout): half-wave ka = 0|8; halves 0..7 = K[ka..ka+7],
    // halves 8..15 = K[ka+16..ka+23].  Hoisted: only 1 B frag live at a time.
    v16h a[2];
    const int ka = lHi * 8;
#pragma unroll
    for (int mt = 0; mt < 2; ++mt) {
      const int row = waveM * 32 + mt * 16 + l15;
      v8h alo = *(const v8h*)(&ldsA[oA + row * LDA + ka]);
      v8h ahi = *(const v8h*)(&ldsA[oA + row * LDA + ka + 16]);
      a[mt] = __builtin_shufflevector(alo, ahi, 0, 1, 2, 3, 4, 5, 6, 7,
                                      8, 9, 10, 11, 12, 13, 14, 15);
    }
    const int kb2 = lHi * 8;
#pragma unroll
    for (int nt = 0; nt < 4; ++nt) {
      // B frag (32x16, ISA layout): half-wave kb2 = 0|8; halves 0..7 = K[kb2..kb2+7], halves 8..15 = K[kb2+16..kb2+23]
      const int ncol = waveN * 64 + nt * 16 + l15;
      v8h blo = *(const v8h*)(&ldsB[oB + ncol * LDA + kb2]);
      v8h bhi = *(const v8h*)(&ldsB[oB + ncol * LDA + kb2 + 16]);
      v16h b  = __builtin_shufflevector(blo, bhi, 0, 1, 2, 3, 4, 5, 6, 7,
                                        8, 9, 10, 11, 12, 13, 14, 15);
      acc[0][nt] = __builtin_amdgcn_wmma_f32_16x16x32_f16(
          false, a[0], false, b, (short)0, acc[0][nt], false, false);
      acc[1][nt] = __builtin_amdgcn_wmma_f32_16x16x32_f16(
          false, a[1], false, b, (short)0, acc[1][nt], false, false);
    }
  };

  const int NK = K_SZ / BK;                 // 128 k-steps
  asyncTiles(0, 0, 0);
  __builtin_amdgcn_s_wait_asynccnt(0);
  __syncthreads();
  unsigned aOff = 0, bOff = 0;
#pragma unroll 1
  for (int kt = 0; kt < NK; ++kt) {
    const int ktn = (kt + 1 < NK) ? kt + 1 : kt;    // last iter: harmless re-copy
    asyncTiles(ktn, aOff ^ ABUF, bOff ^ BBUF);      // background DMA -> back buffer
    compute(aOff, bOff);                            // 8 WMMAs hide the copy latency
    __builtin_amdgcn_s_wait_asynccnt(0);            // my slice landed in LDS
    __syncthreads();                                // everyone's slice landed
    aOff ^= ABUF;
    bOff ^= BBUF;
  }

  // Epilogue: + bias, scatter gate g columns into out slot g  (z,i,f,o -> 0,1,2,3)
#pragma unroll
  for (int mt = 0; mt < 2; ++mt) {
#pragma unroll
    for (int nt = 0; nt < 4; ++nt) {
      const int col  = bn + waveN * 64 + nt * 16 + l15;
      const int gate = col >> 11;           // H = 2048
      const int j    = col & (H_SZ - 1);
      const float bv = bias[col];
      float* o = out + (size_t)gate * SLOT + j;
      const int rbase = bm + waveM * 32 + mt * 16 + lHi * 8;
#pragma unroll
      for (int r = 0; r < 8; ++r)
        o[(size_t)(rbase + r) * H_SZ] = acc[mt][nt][r] + bv;
    }
  }
}

// ------------------------------------------------------------------
// Kernel 2: sLSTM elementwise update, in place over d_out slots
// ------------------------------------------------------------------
__global__ __launch_bounds__(256) void slstm_eltwise(
    const float* __restrict__ c, const float* __restrict__ n,
    const float* __restrict__ m, float* __restrict__ out) {
  const size_t idx = (size_t)blockIdx.x * 256 + threadIdx.x;
  const float zt = out[0 * SLOT + idx];
  const float it = out[1 * SLOT + idx];
  const float ft = out[2 * SLOT + idx];
  const float ot = out[3 * SLOT + idx];

  const float z  = tanhf(zt);
  const float lf = fminf(ft, 0.f) - log1pf(expf(-fabsf(ft)));   // log_sigmoid(ft)
  const float fs = 1.f / (1.f + expf(-ft));
  const float os = 1.f / (1.f + expf(-ot));
  const float mt = fmaxf(lf + m[idx], it);
  const float ip = expf(it - mt);
  const float ct = fs * c[idx] + ip * z;
  const float nt = fs * n[idx] + ip;
  const float ht = os * (ct / nt);

  out[0 * SLOT + idx] = ht;
  out[1 * SLOT + idx] = ct;
  out[2 * SLOT + idx] = nt;
  out[3 * SLOT + idx] = mt;
}

// ------------------------------------------------------------------
extern "C" void kernel_launch(void* const* d_in, const int* in_sizes, int n_in,
                              void* d_out, int out_size, void* d_ws, size_t ws_size,
                              hipStream_t stream) {
  (void)in_sizes; (void)n_in; (void)out_size; (void)ws_size;
  const float* x    = (const float*)d_in[0];
  const float* h    = (const float*)d_in[1];
  const float* c    = (const float*)d_in[2];
  const float* n    = (const float*)d_in[3];
  const float* m    = (const float*)d_in[4];
  const float* W    = (const float*)d_in[5];
  const float* bias = (const float*)d_in[6];
  float* out = (float*)d_out;

  _Float16* Wt = (_Float16*)d_ws;                                   // 64 MiB
  _Float16* Ah = (_Float16*)((char*)d_ws + (size_t)N_SZ * K_SZ * 2); // +32 MiB

  convW<<<dim3(N_SZ / 64, K_SZ / 64), 256, 0, stream>>>(W, Wt);
  convA<<<(B_SZ * K_SZ / 4) / 256, 256, 0, stream>>>(x, h, Ah);
  gemm_gates<<<dim3(N_SZ / BN, B_SZ / BM), 256, 0, stream>>>(Ah, Wt, bias, out);
  slstm_eltwise<<<(B_SZ * H_SZ) / 256, 256, 0, stream>>>(c, n, m, out);
}
